// SAMR_37546604102404
// MI455X (gfx1250) — compile-verified, matrix operand order checked by reading, not run on hardware
//
#include <hip/hip_runtime.h>
#include <hip/hip_bf16.h>

// ---------------------------------------------------------------------------
// GCN + attention pooling for MI455X (gfx1250, wave32, WMMA).
//  * Activations kept fp16 end-to-end: A fragments = 2x b128 loads (the ISA
//    16x32 A layout is two contiguous 16B runs per lane in row-major fp16).
//  * Weights pre-packed per layer into fragment order: B fragment = 32
//    contiguous bytes per lane, zero converts in the hot loop.
//  * Edge gather reads fp16 (halves dominant HBM traffic); scatter uses
//    fp32 global_atomic_add_f32 into an L2-resident accumulator.
// ---------------------------------------------------------------------------

typedef __attribute__((ext_vector_type(16))) _Float16 v16h;
typedef __attribute__((ext_vector_type(8)))  _Float16 v8h;
typedef __attribute__((ext_vector_type(4)))  _Float16 v4h;
typedef __attribute__((ext_vector_type(8)))  float    v8f;

#define DIN  128
#define FD   128
#define DOUT 256

// ---- generic zero --------------------------------------------------------
__global__ void k_zero(float* __restrict__ p, long long n) {
  long long i = (long long)blockIdx.x * blockDim.x + threadIdx.x;
  long long stride = (long long)gridDim.x * blockDim.x;
  for (; i < n; i += stride) p[i] = 0.0f;
}

// ---- fp32 -> fp16 elementwise -------------------------------------------
__global__ void k_cvt_h(const float* __restrict__ s, _Float16* __restrict__ d, long long n) {
  long long i = (long long)blockIdx.x * blockDim.x + threadIdx.x;
  if (i < n) d[i] = (_Float16)s[i];
}

// ---- degree count (self-loop folded in later as +1) ----------------------
__global__ void k_deg(const long long* __restrict__ dst, float* __restrict__ deg, int E) {
  int e = blockIdx.x * blockDim.x + threadIdx.x;
  if (e < E) atomicAdd(&deg[(int)dst[e]], 1.0f);
}

__global__ void k_dinv(const float* __restrict__ deg, float* __restrict__ dinv, int n) {
  int i = blockIdx.x * blockDim.x + threadIdx.x;
  if (i < n) dinv[i] = rsqrtf(deg[i] + 1.0f);   // +1 = self loop; always > 0
}

__global__ void k_enorm(const long long* __restrict__ src, const long long* __restrict__ dst,
                        const float* __restrict__ dinv, float* __restrict__ enorm, int E) {
  int e = blockIdx.x * blockDim.x + threadIdx.x;
  if (e < E) enorm[e] = dinv[(int)src[e]] * dinv[(int)dst[e]];
}

// ---- weight pack: W[K,Nf] fp32 -> fragment-ordered fp16 ------------------
// Bp[((tcol*(K/32)+c)*32+lane)*16 + j] = W[k(c,lane,j)][tcol*16 + lane&15]
__global__ void k_packB(const float* __restrict__ W, _Float16* __restrict__ Bp,
                        int K, int Nf) {
  int idx = blockIdx.x * blockDim.x + threadIdx.x;
  if (idx >= K * Nf) return;
  const int j    = idx & 15;
  const int lane = (idx >> 4) & 31;
  const int rest = idx >> 9;
  const int nch  = K >> 5;
  const int c    = rest % nch;
  const int tcol = rest / nch;
  const int kbase = (lane < 16) ? 0 : 8;
  const int k = c * 32 + ((j < 8) ? (kbase + j) : (16 + kbase + (j - 8)));
  const int n = tcol * 16 + (lane & 15);
  Bp[idx] = (_Float16)W[(size_t)k * Nf + n];
}

// ---- WMMA GEMM: C[M,Nf] = Ah[M,KC] @ Wpacked (+bias)(+relu) --------------
// One wave (32-thread block) computes a 16 x (16*NT) strip; K fully unrolled.
template <int KC, int NT>
__global__ void k_gemm_h(const _Float16* __restrict__ Ah, const _Float16* __restrict__ Bp,
                         const float* __restrict__ bias, float* __restrict__ C32,
                         _Float16* __restrict__ C16, int Nf, int relu) {
  const int lane   = threadIdx.x & 31;
  const int row    = blockIdx.x * 16;
  const int col0   = blockIdx.y * (16 * NT);
  const int l15    = lane & 31 & 15;
  const int kbase  = (lane < 16) ? 0 : 8;

  v8f acc[NT];
#pragma unroll
  for (int t = 0; t < NT; ++t) acc[t] = (v8f){};

  const _Float16* arow = Ah + (size_t)(row + l15) * KC;

#pragma unroll
  for (int c = 0; c < KC / 32; ++c) {
    // A fragment: two contiguous 16B runs per lane (ISA 16x32 fp16 layout)
    const v8h alo = *(const v8h*)(arow + c * 32 + kbase);
    const v8h ahi = *(const v8h*)(arow + c * 32 + 16 + kbase);
    v16h a;
#pragma unroll
    for (int i = 0; i < 8; ++i) { a[i] = alo[i]; a[8 + i] = ahi[i]; }

#pragma unroll
    for (int t = 0; t < NT; ++t) {
      const int tcol = (col0 >> 4) + t;
      const v16h b = *(const v16h*)(Bp + ((size_t)(tcol * (KC / 32) + c) * 32 + lane) * 16);
      acc[t] = __builtin_amdgcn_wmma_f32_16x16x32_f16(false, a, false, b,
                                                      (short)0, acc[t], false, false);
    }
  }

  const int mbase = (lane < 16) ? 0 : 8;     // C/D layout: VGPR v -> row v / 8+v
#pragma unroll
  for (int t = 0; t < NT; ++t) {
    const int   ncol = col0 + t * 16 + l15;
    const float bv   = bias ? bias[ncol] : 0.0f;
#pragma unroll
    for (int v = 0; v < 8; ++v) {
      float o = acc[t][v] + bv;
      if (relu) o = fmaxf(o, 0.0f);
      const size_t idx = (size_t)(row + mbase + v) * Nf + ncol;
      if (C16) C16[idx] = (_Float16)o;
      if (C32) C32[idx] = o;
    }
  }
}

// ---- edge scatter: agg[dst] += h16[src]*norm (fp16 gather, fp32 atomics) -
__global__ void k_scatter4h(const _Float16* __restrict__ h,
                            const long long* __restrict__ src,
                            const long long* __restrict__ dst,
                            const float* __restrict__ enorm,
                            float* __restrict__ agg, int E, int F) {
  const int fpr = F >> 2;
  long long t = (long long)blockIdx.x * blockDim.x + threadIdx.x;
  if (t >= (long long)E * fpr) return;
  const int e  = (int)(t / fpr);
  const int fo = (int)(t % fpr) * 4;
  const float nrm = enorm[e];
  const v4h v = *(const v4h*)(h + (size_t)src[e] * F + fo);
  float* o = agg + (size_t)dst[e] * F + fo;
  atomicAdd(o + 0, (float)v[0] * nrm);
  atomicAdd(o + 1, (float)v[1] * nrm);
  atomicAdd(o + 2, (float)v[2] * nrm);
  atomicAdd(o + 3, (float)v[3] * nrm);
}

__global__ void k_scatter1(const float* __restrict__ h,
                           const long long* __restrict__ src,
                           const long long* __restrict__ dst,
                           const float* __restrict__ enorm,
                           float* __restrict__ agg, int E) {
  int e = blockIdx.x * blockDim.x + threadIdx.x;
  if (e < E) atomicAdd(&agg[(int)dst[e]], h[(int)src[e]] * enorm[e]);
}

// ---- finalize: out16 = relu(agg + h16*dinv^2 + bias) ---------------------
__global__ void k_fin_h(const float* __restrict__ agg, const _Float16* __restrict__ h,
                        const float* __restrict__ dinv, const float* __restrict__ bias,
                        _Float16* __restrict__ out, int n, int F) {
  long long t = (long long)blockIdx.x * blockDim.x + threadIdx.x;
  if (t >= (long long)n * F) return;
  const int i = (int)(t / F);
  const int f = (int)(t % F);
  const float d = dinv[i];
  const float v = agg[t] + (float)h[t] * d * d + bias[f];
  out[t] = (_Float16)fmaxf(v, 0.0f);
}

// scalar (F==1, fp32 in/out) for the attention head
__global__ void k_fin1(const float* __restrict__ agg, const float* __restrict__ h,
                       const float* __restrict__ dinv, const float* __restrict__ bias,
                       float* __restrict__ out, int n) {
  int i = blockIdx.x * blockDim.x + threadIdx.x;
  if (i < n) {
    const float d = dinv[i];
    out[i] = fmaxf(agg[i] + h[i] * d * d + bias[0], 0.0f);
  }
}

// ---- K=32, Nf=1 head: wave-per-node dot product (fp16 input) -------------
__global__ void k_dot32h(const _Float16* __restrict__ A, const float* __restrict__ w,
                         float* __restrict__ out, int n) {
  const int i    = blockIdx.x * (blockDim.x >> 5) + (threadIdx.x >> 5);
  const int lane = threadIdx.x & 31;
  if (i >= n) return;
  float v = (float)A[(size_t)i * 32 + lane] * w[lane];
#pragma unroll
  for (int off = 16; off; off >>= 1) v += __shfl_xor(v, off);
  if (lane == 0) out[i] = v;
}

// ---- segment softmax pieces (w >= 0 after ReLU -> uint-bit max valid) ----
__global__ void k_segmax(const float* __restrict__ w, const long long* __restrict__ batch,
                         float* __restrict__ gmax, int n) {
  int i = blockIdx.x * blockDim.x + threadIdx.x;
  if (i < n) atomicMax((unsigned int*)&gmax[(int)batch[i]], __float_as_uint(w[i]));
}

__global__ void k_expsum(const float* __restrict__ w, const long long* __restrict__ batch,
                         const float* __restrict__ gmax, float* __restrict__ ev,
                         float* __restrict__ gsum, int n) {
  int i = blockIdx.x * blockDim.x + threadIdx.x;
  if (i < n) {
    const int g = (int)batch[i];
    const float e = expf(w[i] - gmax[g]);
    ev[i] = e;
    atomicAdd(&gsum[g], e);
  }
}

// ---- pooling: pooled[g] += f16[i]*ev[i]/(gsum[g]+eps) --------------------
__global__ void k_pool4h(const _Float16* __restrict__ f, const float* __restrict__ ev,
                         const float* __restrict__ gsum, const long long* __restrict__ batch,
                         float* __restrict__ pooled, int n, int F) {
  const int fpr = F >> 2;
  long long t = (long long)blockIdx.x * blockDim.x + threadIdx.x;
  if (t >= (long long)n * fpr) return;
  const int i  = (int)(t / fpr);
  const int fo = (int)(t % fpr) * 4;
  const int g  = (int)batch[i];
  const float c = ev[i] / (gsum[g] + 1e-16f);
  const v4h v = *(const v4h*)(f + (size_t)i * F + fo);
  float* o = pooled + (size_t)g * F + fo;
  atomicAdd(o + 0, (float)v[0] * c);
  atomicAdd(o + 1, (float)v[1] * c);
  atomicAdd(o + 2, (float)v[2] * c);
  atomicAdd(o + 3, (float)v[3] * c);
}

// ---------------------------------------------------------------------------
static inline int cdiv_ll(long long a, long long b) { return (int)((a + b - 1) / b); }

static void gemm_packed(const float* W, const _Float16* Ah, const float* bias,
                        float* C32, _Float16* C16, int M, int K, int Nf, int relu,
                        _Float16* Bp, hipStream_t s) {
  k_packB<<<cdiv_ll((long long)K * Nf, 256), 256, 0, s>>>(W, Bp, K, Nf);
  const bool wide = (Nf % 64) == 0;
  dim3 grid(M / 16, wide ? Nf / 64 : Nf / 32);
  if (K == 128) {
    if (wide) k_gemm_h<128, 4><<<grid, 32, 0, s>>>(Ah, Bp, bias, C32, C16, Nf, relu);
    else      k_gemm_h<128, 2><<<grid, 32, 0, s>>>(Ah, Bp, bias, C32, C16, Nf, relu);
  } else if (K == 64) {
    if (wide) k_gemm_h<64, 4><<<grid, 32, 0, s>>>(Ah, Bp, bias, C32, C16, Nf, relu);
    else      k_gemm_h<64, 2><<<grid, 32, 0, s>>>(Ah, Bp, bias, C32, C16, Nf, relu);
  } else {
    if (wide) k_gemm_h<32, 4><<<grid, 32, 0, s>>>(Ah, Bp, bias, C32, C16, Nf, relu);
    else      k_gemm_h<32, 2><<<grid, 32, 0, s>>>(Ah, Bp, bias, C32, C16, Nf, relu);
  }
}

extern "C" void kernel_launch(void* const* d_in, const int* in_sizes, int n_in,
                              void* d_out, int out_size, void* d_ws, size_t ws_size,
                              hipStream_t stream) {
  const float*     x     = (const float*)d_in[0];
  const long long* eidx  = (const long long*)d_in[1];
  const long long* batch = (const long long*)d_in[2];
  const float *W1 = (const float*)d_in[3],  *b1  = (const float*)d_in[4];
  const float *W2 = (const float*)d_in[5],  *b2  = (const float*)d_in[6];
  const float *W3 = (const float*)d_in[7],  *b3  = (const float*)d_in[8];
  const float *A1 = (const float*)d_in[9],  *ba1 = (const float*)d_in[10];
  const float *A2 = (const float*)d_in[11], *ba2 = (const float*)d_in[12];
  const float *A3 = (const float*)d_in[13], *ba3 = (const float*)d_in[14];
  const float *M1 = (const float*)d_in[15], *bm1 = (const float*)d_in[16];
  const float *M2 = (const float*)d_in[17], *bm2 = (const float*)d_in[18];

  const int N = in_sizes[0] / DIN;       // 50000 (multiple of 16)
  const int E = in_sizes[1] / 2;         // 800000
  const int G = out_size / DOUT;         // 64
  const long long* src = eidx;
  const long long* dst = eidx + E;

  // workspace layout (in float units; fp16 buffers use half the floats)
  float* p = (float*)d_ws;
  float*    deg    = p;               p += N;
  float*    dinv   = p;               p += N;
  float*    enorm  = p;               p += E;
  float*    aggbuf = p;               p += (size_t)N * FD;        // fp32 scatter acc
  _Float16* xh     = (_Float16*)p;    p += (size_t)N * (FD / 2);  // x in fp16
  _Float16* hbuf   = (_Float16*)p;    p += (size_t)N * (FD / 2);  // GEMM out (fp16)
  _Float16* fbuf   = (_Float16*)p;    p += (size_t)N * (FD / 2);  // layer out (fp16)
  _Float16* aH     = (_Float16*)p;    p += (size_t)N * 32;        // attn GEMM out
  _Float16* aF     = (_Float16*)p;    p += (size_t)N * 32;        // attn layer out
  float*    wlin   = p;               p += N;
  float*    wagg   = p;               p += N;
  float*    wv     = p;               p += N;
  float*    ev     = p;               p += N;
  float*    gmax   = p;               p += G;
  float*    gsum   = p;               p += G;
  float*    pooled = p;               p += (size_t)G * FD;
  _Float16* pooledh= (_Float16*)p;    p += (size_t)G * (FD / 2);
  _Float16* hidden = (_Float16*)p;    p += (size_t)G * (FD / 2);
  _Float16* Bp     = (_Float16*)p;    p += (128 * 256) / 2;       // packed weights

  auto zero = [&](float* q, long long n) {
    int blocks = cdiv_ll(n, 256); if (blocks > 4096) blocks = 4096;
    k_zero<<<blocks, 256, 0, stream>>>(q, n);
  };
  auto gcn_layer = [&](const _Float16* in, const float* W, const float* bias,
                       _Float16* H, float* AGG, _Float16* OUT, int K, int F) {
    gemm_packed(W, in, nullptr, nullptr, H, N, K, F, 0, Bp, stream);
    zero(AGG, (long long)N * F);
    long long tot = (long long)E * (F >> 2);
    k_scatter4h<<<cdiv_ll(tot, 256), 256, 0, stream>>>(H, src, dst, enorm, AGG, E, F);
    k_fin_h<<<cdiv_ll((long long)N * F, 256), 256, 0, stream>>>(AGG, H, dinv, bias,
                                                                OUT, N, F);
  };

  // ---- normalization coefficients + fp16 copy of x ----
  zero(deg, N);
  k_deg<<<cdiv_ll(E, 256), 256, 0, stream>>>(dst, deg, E);
  k_dinv<<<cdiv_ll(N, 256), 256, 0, stream>>>(deg, dinv, N);
  k_enorm<<<cdiv_ll(E, 256), 256, 0, stream>>>(src, dst, dinv, enorm, E);
  k_cvt_h<<<cdiv_ll((long long)N * DIN, 256), 256, 0, stream>>>(x, xh, (long long)N * DIN);

  // ---- feature branch: 3x GCN(128->128) + ReLU ----
  gcn_layer(xh,   W1, b1, hbuf, aggbuf, fbuf, DIN, FD);
  gcn_layer(fbuf, W2, b2, hbuf, aggbuf, fbuf, FD,  FD);
  gcn_layer(fbuf, W3, b3, hbuf, aggbuf, fbuf, FD,  FD);

  // ---- attention branch: GCN(128->64), GCN(64->32), GCN(32->1) ----
  gcn_layer(xh, A1, ba1, aH, aggbuf, aF, DIN, 64);
  gcn_layer(aF, A2, ba2, aH, aggbuf, aF, 64,  32);
  // last layer: Nf=1 head via wave dot product, then scalar scatter
  k_dot32h<<<cdiv_ll(N, 8), 256, 0, stream>>>(aF, A3, wlin, N);
  zero(wagg, N);
  k_scatter1<<<cdiv_ll(E, 256), 256, 0, stream>>>(wlin, src, dst, enorm, wagg, E);
  k_fin1<<<cdiv_ll(N, 256), 256, 0, stream>>>(wagg, wlin, dinv, ba3, wv, N);

  // ---- per-graph softmax (w >= 0) ----
  zero(gmax, G);
  zero(gsum, G);
  k_segmax<<<cdiv_ll(N, 256), 256, 0, stream>>>(wv, batch, gmax, N);
  k_expsum<<<cdiv_ll(N, 256), 256, 0, stream>>>(wv, batch, gmax, ev, gsum, N);

  // ---- attention-weighted pooling ----
  zero(pooled, (long long)G * FD);
  long long ptot = (long long)N * (FD >> 2);
  k_pool4h<<<cdiv_ll(ptot, 256), 256, 0, stream>>>(fbuf, ev, gsum, batch, pooled, N, FD);

  // ---- MLP head: relu(pooled@M1+bm1) @ M2 + bm2 ----
  k_cvt_h<<<cdiv_ll((long long)G * FD, 256), 256, 0, stream>>>(pooled, pooledh,
                                                               (long long)G * FD);
  gemm_packed(M1, pooledh, bm1, nullptr, hidden, G, FD, FD, 1, Bp, stream);
  gemm_packed(M2, hidden,  bm2, (float*)d_out, nullptr, G, FD, DOUT, 0, Bp, stream);
}
